// InvertedResidualQ_51479478010587
// MI455X (gfx1250) — matrix operand order repaired and checked by reading, not compiled, not run on hardware
//
#include <hip/hip_runtime.h>

typedef _Float16 half_t;
typedef __attribute__((ext_vector_type(16))) _Float16 v16h;
typedef __attribute__((ext_vector_type(8)))  _Float16 v8h;
typedef __attribute__((ext_vector_type(8)))  float    v8f;

#define M_TOT 401408   // N*H*W = 32*112*112
#define HW    12544    // 112*112
#define WIDTH 112
#define CIN   32
#define CHID  192

// ---------------- utility: wave32 max reduction ----------------
__device__ inline float wave_max(float v) {
  #pragma unroll
  for (int off = 16; off > 0; off >>= 1)
    v = fmaxf(v, __shfl_xor(v, off, 32));
  return v;
}

// ---------------- kernel 0: zero the scalar slab ----------------
__global__ void k_zero(unsigned* scu) {
  if (threadIdx.x < 16) scu[threadIdx.x] = 0u;
}

// ---------------- kernel 1: global max|x| ----------------
__global__ __launch_bounds__(256) void k_absmax(const float* __restrict__ x, size_t n,
                                                unsigned* __restrict__ scu, int slot) {
  __shared__ float red[8];
  float v = 0.f;
  for (size_t i = (size_t)blockIdx.x * 256 + threadIdx.x; i < n; i += (size_t)gridDim.x * 256)
    v = fmaxf(v, fabsf(x[i]));
  v = wave_max(v);
  int wave = threadIdx.x >> 5, lane = threadIdx.x & 31;
  if (lane == 0) red[wave] = v;
  __syncthreads();
  if (threadIdx.x == 0) {
    float m = red[0];
    #pragma unroll
    for (int i = 1; i < 8; ++i) m = fmaxf(m, red[i]);
    atomicMax(scu + slot, __float_as_uint(m));
  }
}

// ---------------- kernel 2: weight codes + BN folds ----------------
__device__ float block_absmax(const float* p, int n, float* red) {
  float v = 0.f;
  for (int i = threadIdx.x; i < n; i += 256) v = fmaxf(v, fabsf(p[i]));
  red[threadIdx.x] = v;
  __syncthreads();
  for (int s = 128; s > 0; s >>= 1) {
    if (threadIdx.x < s) red[threadIdx.x] = fmaxf(red[threadIdx.x], red[threadIdx.x + s]);
    __syncthreads();
  }
  float r = red[0];
  __syncthreads();
  return r;
}

__global__ __launch_bounds__(256) void k_prep(
    const float* __restrict__ w1, const float* __restrict__ wdw, const float* __restrict__ w2,
    const float* __restrict__ g1, const float* __restrict__ b1,
    const float* __restrict__ m1, const float* __restrict__ v1,
    const float* __restrict__ g2, const float* __restrict__ b2,
    const float* __restrict__ m2, const float* __restrict__ v2,
    const float* __restrict__ g3, const float* __restrict__ b3,
    const float* __restrict__ m3, const float* __restrict__ v3,
    float* __restrict__ scf,
    half_t* __restrict__ w1c, half_t* __restrict__ wdwc, half_t* __restrict__ w2c,
    float* __restrict__ bn1a, float* __restrict__ bn1b,
    float* __restrict__ bn2a, float* __restrict__ bn2b,
    float* __restrict__ bn3a, float* __restrict__ bn3b) {
  __shared__ float red[256];
  float s1 = block_absmax(w1,  CHID * CIN, red) + 1e-8f;
  float sd = block_absmax(wdw, CHID * 9,   red) + 1e-8f;
  float s2 = block_absmax(w2,  CIN * CHID, red) + 1e-8f;
  if (threadIdx.x == 0) { scf[4] = s1; scf[5] = sd; scf[6] = s2; }
  for (int i = threadIdx.x; i < CHID * CIN; i += 256)
    w1c[i] = (half_t)rintf(fminf(fmaxf(w1[i] * (7.f / s1), -7.f), 7.f));
  for (int i = threadIdx.x; i < CHID * 9; i += 256)
    wdwc[i] = (half_t)rintf(fminf(fmaxf(wdw[i] * (7.f / sd), -7.f), 7.f));
  for (int i = threadIdx.x; i < CIN * CHID; i += 256)
    w2c[i] = (half_t)rintf(fminf(fmaxf(w2[i] * (7.f / s2), -7.f), 7.f));
  // fold BN: a = g * rsqrt(v+eps), b' = b - m*a (per channel, once)
  if (threadIdx.x < CHID) {
    int c = threadIdx.x;
    float a1 = g1[c] * rsqrtf(v1[c] + 1e-5f);
    bn1a[c] = a1; bn1b[c] = b1[c] - m1[c] * a1;
    float a2 = g2[c] * rsqrtf(v2[c] + 1e-5f);
    bn2a[c] = a2; bn2b[c] = b2[c] - m2[c] * a2;
    if (c < CIN) {
      float a3 = g3[c] * rsqrtf(v3[c] + 1e-5f);
      bn3a[c] = a3; bn3b[c] = b3[c] - m3[c] * a3;
    }
  }
}

// ---------------- kernel 3: expand 1x1 (WMMA GEMM, K=32) + BN1 + ReLU6 + max --------
// Each wave: 32 spatial columns (2 B matrices), 12 channel tiles -> 24 WMMAs.
__global__ __launch_bounds__(256) void k_expand(
    const float* __restrict__ x, const half_t* __restrict__ w1c,
    const float* __restrict__ bn1a, const float* __restrict__ bn1b,
    const float* __restrict__ scf, unsigned* __restrict__ scu,
    half_t* __restrict__ h1)
{
  int wave = threadIdx.x >> 5, lane = threadIdx.x & 31;
  int nlo = lane & 15, kHalf = lane >> 4;
  int col0 = blockIdx.x * 256 + wave * 32 + nlo;
  int col1 = col0 + 16;

  float sx  = scf[0] + 1e-8f;   // act scale (signed, n=7)
  float sw1 = scf[4];           // weight scale
  float qs = 7.f / sx;
  int kBase = kHalf * 16;

  // B matrices: lane holds one column, K packed 2 per VGPR
  v16h bmat0, bmat1;
  {
    int n0 = col0 / HW, hw0 = col0 % HW;
    const float* xp0 = x + (size_t)n0 * CIN * HW + hw0;
    int n1 = col1 / HW, hw1 = col1 % HW;
    const float* xp1 = x + (size_t)n1 * CIN * HW + hw1;
    #pragma unroll
    for (int i = 0; i < 16; ++i) {
      float a = xp0[(size_t)(kBase + i) * HW];
      float b = xp1[(size_t)(kBase + i) * HW];
      bmat0[i] = (half_t)rintf(fminf(fmaxf(a * qs, -7.f), 7.f));
      bmat1[i] = (half_t)rintf(fminf(fmaxf(b * qs, -7.f), 7.f));
    }
  }

  float gs = sx * sw1 * (1.f / 49.f);
  float lmax = 0.f;
  #pragma unroll 1
  for (int ct = 0; ct < 12; ++ct) {               // 12 * 16 = 192 output channels
    int row = ct * 16 + nlo;
    const half_t* ap = w1c + row * 32 + kHalf * 16;
    v8h alo = *(const v8h*)ap;                    // K = kHalf*16 .. +7
    v8h ahi = *(const v8h*)(ap + 8);              // K = kHalf*16 + 8 ..
    v16h amat;
    #pragma unroll
    for (int i = 0; i < 8; ++i) { amat[i] = alo[i]; amat[8 + i] = ahi[i]; }
    v8f acc0 = {}, acc1 = {};
    acc0 = __builtin_amdgcn_wmma_f32_16x16x32_f16(false, amat, false, bmat0,
                                                  (short)0, acc0, false, false);
    acc1 = __builtin_amdgcn_wmma_f32_16x16x32_f16(false, amat, false, bmat1,
                                                  (short)0, acc1, false, false);
    #pragma unroll
    for (int r = 0; r < 8; ++r) {
      int c = ct * 16 + r + kHalf * 8;
      float a = gs * bn1a[c], b = bn1b[c];
      float y0 = fminf(fmaxf(fmaf(acc0[r], a, b), 0.f), 6.f);
      float y1 = fminf(fmaxf(fmaf(acc1[r], a, b), 0.f), 6.f);
      lmax = fmaxf(lmax, fmaxf(y0, y1));
      h1[(size_t)c * M_TOT + col0] = (half_t)y0;
      h1[(size_t)c * M_TOT + col1] = (half_t)y1;
    }
  }
  lmax = wave_max(lmax);
  if (lane == 0) atomicMax(scu + 1, __float_as_uint(lmax));
}

// ---------------- kernel 4: depthwise 3x3 + BN2 + ReLU6 + max ----------------
__global__ __launch_bounds__(256) void k_dw(
    const half_t* __restrict__ h1, const half_t* __restrict__ wdwc,
    const float* __restrict__ bn2a, const float* __restrict__ bn2b,
    const float* __restrict__ scf, unsigned* __restrict__ scu,
    half_t* __restrict__ h2)
{
  __shared__ float red[8];
  size_t tid = (size_t)blockIdx.x * 256 + threadIdx.x;
  int c = (int)(tid / M_TOT);
  int m = (int)(tid % M_TOT);
  float s1 = scf[1] + 1e-8f, sd = scf[5];
  float qs = 15.f / s1;                       // unsigned act-quant, n=15
  int hw = m % HW, h = hw / WIDTH, w = hw % WIDTH;
  const half_t* base = h1 + (size_t)c * M_TOT + (size_t)(m - hw);
  float acc = 0.f;
  #pragma unroll
  for (int dh = -1; dh <= 1; ++dh)
    #pragma unroll
    for (int dw = -1; dw <= 1; ++dw) {
      int hh = h + dh, ww = w + dw;
      float q = 0.f;
      if (hh >= 0 && hh < WIDTH && ww >= 0 && ww < WIDTH) {
        float v = (float)base[hh * WIDTH + ww];
        q = rintf(fminf(v * qs, 15.f));       // v >= 0 after ReLU6
      }
      acc += q * (float)wdwc[c * 9 + (dh + 1) * 3 + (dw + 1)];
    }
  float gsd = (s1 * sd) * (1.f / 105.f);      // (s1/15)*(sd/7)
  float y = fmaf(acc, gsd * bn2a[c], bn2b[c]);
  y = fminf(fmaxf(y, 0.f), 6.f);
  h2[tid] = (half_t)y;
  float lmax = wave_max(y);
  int wave = threadIdx.x >> 5, lane = threadIdx.x & 31;
  if (lane == 0) red[wave] = lmax;
  __syncthreads();
  if (threadIdx.x == 0) {
    float mx = red[0];
    #pragma unroll
    for (int i = 1; i < 8; ++i) mx = fmaxf(mx, red[i]);
    atomicMax(scu + 2, __float_as_uint(mx));
  }
}

// ---------------- kernel 5: project 1x1 (WMMA GEMM, K=192) + BN3 + |max| ----------------
__global__ __launch_bounds__(256) void k_project(
    const half_t* __restrict__ h2, const half_t* __restrict__ w2c,
    const float* __restrict__ bn3a, const float* __restrict__ bn3b,
    const float* __restrict__ scf, unsigned* __restrict__ scu,
    float* __restrict__ h3)
{
  int wave = threadIdx.x >> 5, lane = threadIdx.x & 31;
  int nlo = lane & 15, kHalf = lane >> 4;
  int col = blockIdx.x * 128 + wave * 16 + nlo;
  float s2 = scf[2] + 1e-8f, sw2 = scf[6];
  float qs = 15.f / s2;
  int kBase = kHalf * 16;

  v8f acc0 = {}, acc1 = {};                 // output channels 0-15 and 16-31
  #pragma unroll 1
  for (int kk = 0; kk < 6; ++kk) {          // K = 192 in 6 chunks of 32
    if (kk < 5)                             // prefetch next K-chunk of B operand
      __builtin_prefetch(&h2[(size_t)(kk * 32 + 32 + kBase) * M_TOT + col], 0, 1);
    v16h bmat;
    #pragma unroll
    for (int i = 0; i < 16; ++i) {
      int c = kk * 32 + kBase + i;
      float v = (float)h2[(size_t)c * M_TOT + col];
      bmat[i] = (half_t)rintf(fminf(v * qs, 15.f));
    }
    const half_t* ap0 = w2c + nlo * CHID + kk * 32 + kHalf * 16;
    const half_t* ap1 = w2c + (16 + nlo) * CHID + kk * 32 + kHalf * 16;
    v8h a0lo = *(const v8h*)ap0, a0hi = *(const v8h*)(ap0 + 8);
    v8h a1lo = *(const v8h*)ap1, a1hi = *(const v8h*)(ap1 + 8);
    v16h am0, am1;
    #pragma unroll
    for (int i = 0; i < 8; ++i) {
      am0[i] = a0lo[i]; am0[8 + i] = a0hi[i];
      am1[i] = a1lo[i]; am1[8 + i] = a1hi[i];
    }
    acc0 = __builtin_amdgcn_wmma_f32_16x16x32_f16(false, am0, false, bmat,
                                                  (short)0, acc0, false, false);
    acc1 = __builtin_amdgcn_wmma_f32_16x16x32_f16(false, am1, false, bmat,
                                                  (short)0, acc1, false, false);
  }
  float gs = (s2 * sw2) * (1.f / 105.f);
  float lmax = 0.f;
  #pragma unroll
  for (int r = 0; r < 8; ++r) {
    int c0 = r + kHalf * 8;
    float y0 = fmaf(acc0[r], gs * bn3a[c0], bn3b[c0]);
    lmax = fmaxf(lmax, fabsf(y0));
    h3[(size_t)c0 * M_TOT + col] = y0;
    int c1 = c0 + 16;
    float y1 = fmaf(acc1[r], gs * bn3a[c1], bn3b[c1]);
    lmax = fmaxf(lmax, fabsf(y1));
    h3[(size_t)c1 * M_TOT + col] = y1;
  }
  lmax = wave_max(lmax);
  if (lane == 0) atomicMax(scu + 3, __float_as_uint(lmax));
}

// ---------------- kernel 6: signed act-quant of h3 + residual add ----------------
__global__ __launch_bounds__(256) void k_final(
    const float* __restrict__ x, const float* __restrict__ h3,
    const float* __restrict__ scf, float* __restrict__ out)
{
  size_t i = (size_t)blockIdx.x * 256 + threadIdx.x;
  int hw = (int)(i % HW);
  int c  = (int)((i / HW) % CIN);
  int n  = (int)(i / ((size_t)HW * CIN));
  float s3 = scf[3] + 1e-8f;
  float v = h3[(size_t)c * M_TOT + (size_t)n * HW + hw];
  float q = rintf(fminf(fmaxf(v * (7.f / s3), -7.f), 7.f));
  out[i] = x[i] + q * (s3 * (1.f / 7.f));
}

extern "C" void kernel_launch(void* const* d_in, const int* in_sizes, int n_in,
                              void* d_out, int out_size, void* d_ws, size_t ws_size,
                              hipStream_t stream) {
  const float* x   = (const float*)d_in[0];
  const float* w1  = (const float*)d_in[1];
  const float* g1  = (const float*)d_in[2];
  const float* b1  = (const float*)d_in[3];
  const float* m1  = (const float*)d_in[4];
  const float* v1  = (const float*)d_in[5];
  const float* wdw = (const float*)d_in[6];
  const float* g2  = (const float*)d_in[7];
  const float* b2  = (const float*)d_in[8];
  const float* m2  = (const float*)d_in[9];
  const float* v2  = (const float*)d_in[10];
  const float* w2  = (const float*)d_in[11];
  const float* g3  = (const float*)d_in[12];
  const float* b3  = (const float*)d_in[13];
  const float* m3  = (const float*)d_in[14];
  const float* v3  = (const float*)d_in[15];
  float* out = (float*)d_out;

  char* ws = (char*)d_ws;
  unsigned* scu = (unsigned*)ws;          // slots 0..3: float-bit maxima (>=0)
  float*    scf = (float*)ws;             // slots 4..6: weight scales
  half_t* w1c  = (half_t*)(ws + 256);     // 192*32 f16 codes
  half_t* w2c  = (half_t*)(ws + 12544);   // 32*192 f16 codes
  half_t* wdwc = (half_t*)(ws + 24832);   // 192*9  f16 codes
  float*  bn1a = (float*)(ws + 28288);    // 192
  float*  bn1b = (float*)(ws + 29056);    // 192
  float*  bn2a = (float*)(ws + 29824);    // 192
  float*  bn2b = (float*)(ws + 30592);    // 192
  float*  bn3a = (float*)(ws + 31360);    // 32
  float*  bn3b = (float*)(ws + 31488);    // 32
  half_t* h1   = (half_t*)(ws + 32768);                      // 192*M f16
  half_t* h2   = (half_t*)(ws + 32768 + 154140672ull);       // 192*M f16
  float*  h3   = (float*) (ws + 32768 + 2ull * 154140672ull);// 32*M  f32

  k_zero<<<1, 32, 0, stream>>>(scu);
  k_absmax<<<1024, 256, 0, stream>>>(x, (size_t)CIN * M_TOT, scu, 0);
  k_prep<<<1, 256, 0, stream>>>(w1, wdw, w2,
                                g1, b1, m1, v1, g2, b2, m2, v2, g3, b3, m3, v3,
                                scf, w1c, wdwc, w2c,
                                bn1a, bn1b, bn2a, bn2b, bn3a, bn3b);
  k_expand<<<M_TOT / 256, 256, 0, stream>>>(x, w1c, bn1a, bn1b, scf, scu, h1);
  k_dw<<<(int)(((size_t)CHID * M_TOT) / 256), 256, 0, stream>>>(h1, wdwc, bn2a, bn2b,
                                                                scf, scu, h2);
  k_project<<<M_TOT / 128, 256, 0, stream>>>(h2, w2c, bn3a, bn3b, scf, scu, h3);
  k_final<<<(int)(((size_t)CIN * M_TOT) / 256), 256, 0, stream>>>(x, h3, scf, out);
}
